// ProteinMPNN_3504693314241
// MI455X (gfx1250) — compile-verified, matrix operand order checked by reading, not run on hardware
//
#include <hip/hip_runtime.h>
#include <hip/hip_bf16.h>

typedef __attribute__((ext_vector_type(16))) _Float16 v16h;
typedef __attribute__((ext_vector_type(8)))  float    v8f;

#define Bn 4
#define Ln 2048
#define Kn 32
#define Hn 128
#define ACOLS 384        // H + NIN
#define FFn 512
#define SCALE_INV (1.0f/30.0f)
#define EPSF 1e-5f

// K-offset of element pair held in "VGPR" j for lanes 0-15 (half=0) / 16-31 (half=1)
// per CDNA5 ISA 16-bit A/B 16x32 fragment layout.
__device__ __forceinline__ int kbase(int j, int half) {
  return (j < 4 ? (j << 1) : (16 + ((j - 4) << 1))) + (half ? 8 : 0);
}

__device__ __forceinline__ v8f wmma_f16(v16h a, v16h b, v8f c) {
  // (neg_a, A, neg_b, B, c_mod, C, reuse_a, reuse_b)
  return __builtin_amdgcn_wmma_f32_16x16x32_f16(false, a, false, b, (short)0, c,
                                                false, false);
}

// Load a 16x32 f16 A fragment from an LDS row-major [rows][ncols] buffer.
__device__ __forceinline__ v16h load_a_frag(const _Float16* buf, int ncols,
                                            int mt, int ks, int lane) {
  int m    = mt * 16 + (lane & 15);
  int half = (lane >> 4) & 1;
  const _Float16* row = buf + m * ncols + ks * 32;
  v16h a;
#pragma unroll
  for (int j = 0; j < 8; ++j) {
    int kb = kbase(j, half);
    a[2 * j]     = row[kb];
    a[2 * j + 1] = row[kb + 1];
  }
  return a;
}

__device__ __forceinline__ float gelu_exact(float x) {
  return 0.5f * x * (1.0f + erff(x * 0.7071067811865475f));
}

// ---------------------------------------------------------------------------
// Pack fp32 row-major W[Kdim][Ndim] into f16 B-fragment order:
// dst[((kt*nN + nb)*32 + lane)*16 + 2*j + p] = W[kt*32 + kbase(j,half) + p][nb*16 + lane%16]
// so each lane reads its whole WMMA B operand as one contiguous 32B load.
// ---------------------------------------------------------------------------
__global__ __launch_bounds__(256) void pack_weight(const float* __restrict__ src,
                                                   _Float16* __restrict__ dst,
                                                   int Kdim, int Ndim) {
  int idx = blockIdx.x * 256 + threadIdx.x;
  if (idx >= Kdim * Ndim) return;
  int p    = idx & 1;
  int j    = (idx >> 1) & 7;
  int lane = (idx >> 4) & 31;
  int tile = idx >> 9;
  int nN   = Ndim >> 4;
  int nb   = tile % nN;
  int kt   = tile / nN;
  int half = lane >> 4;
  int k    = kt * 32 + kbase(j, half) + p;
  int n    = nb * 16 + (lane & 15);
  dst[idx] = (_Float16)src[k * Ndim + n];
}

// ---------------------------------------------------------------------------
// Kernel 1: per-(b,l) node message MLP over K=32 neighbors, masked K-sum,
// residual + LN1.  One 256-thread WG (8 waves) per node.
// ---------------------------------------------------------------------------
__global__ __launch_bounds__(256) void node_msg_ln1(
    const float* __restrict__ hV, const float* __restrict__ hE,
    const int* __restrict__ Eidx, const float* __restrict__ maskA,
    const _Float16* __restrict__ W1p, const float* __restrict__ b1,
    const _Float16* __restrict__ W2p, const float* __restrict__ b2,
    const _Float16* __restrict__ W3p, const float* __restrict__ b3,
    const float* __restrict__ ln1g, const float* __restrict__ ln1b,
    float* __restrict__ hV1) {
  __shared__ _Float16 abuf[32 * ACOLS];   // 24 KB  concat activations (f16)
  __shared__ _Float16 mbuf[32 * Hn];      //  8 KB  inter-layer activations
  __shared__ float colsum[Hn];
  __shared__ float red[Hn], red2[Hn];

  const int bl   = blockIdx.x;            // b*L + l
  const int b    = bl >> 11;              // / 2048
  const int tid  = threadIdx.x;
  const int lane = tid & 31, wave = tid >> 5;
  const v8f vzero = {0, 0, 0, 0, 0, 0, 0, 0};

  const float* hv_row = hV + (size_t)bl * Hn;
  const float* he_row = hE + (size_t)bl * Kn * Hn;
  const int*   ix_row = Eidx + (size_t)bl * Kn;

  // Build A = [h_V | h_E | h_V[idx]]  (32 rows x 384 cols, f16)
  for (int e = tid; e < 32 * ACOLS; e += 256) {
    int r = e / ACOLS, c = e - r * ACOLS;
    float v;
    if (c < Hn)            v = hv_row[c];
    else if (c < 2 * Hn)   v = he_row[r * Hn + (c - Hn)];
    else                   v = hV[((size_t)b * Ln + ix_row[r]) * Hn + (c - 2 * Hn)];
    abuf[e] = (_Float16)v;
  }
  if (tid < Hn) colsum[tid] = 0.0f;
  __syncthreads();

  const int nb = wave;                    // wave owns N-block nb, both M tiles
  const int n  = nb * 16 + (lane & 15);
  const int mh = (lane >> 4) ? 8 : 0;

  // ---- layer 1 : 32x384 @ 384x128 ----
  v8f c0 = vzero, c1 = vzero;
  for (int ks = 0; ks < 12; ++ks) {
    v16h bf = *(const v16h*)(W1p + ((size_t)(ks * 8 + nb) << 9) + (lane << 4));
    c0 = wmma_f16(load_a_frag(abuf, ACOLS, 0, ks, lane), bf, c0);
    c1 = wmma_f16(load_a_frag(abuf, ACOLS, 1, ks, lane), bf, c1);
  }
  {
    float bb = b1[n];
#pragma unroll
    for (int r = 0; r < 8; ++r) {
      mbuf[(mh + r) * Hn + n]      = (_Float16)gelu_exact(c0[r] + bb);
      mbuf[(16 + mh + r) * Hn + n] = (_Float16)gelu_exact(c1[r] + bb);
    }
  }
  __syncthreads();

  // ---- layer 2 : 32x128 @ 128x128 ----
  v8f d0 = vzero, d1 = vzero;
  for (int ks = 0; ks < 4; ++ks) {
    v16h bf = *(const v16h*)(W2p + ((size_t)(ks * 8 + nb) << 9) + (lane << 4));
    d0 = wmma_f16(load_a_frag(mbuf, Hn, 0, ks, lane), bf, d0);
    d1 = wmma_f16(load_a_frag(mbuf, Hn, 1, ks, lane), bf, d1);
  }
  __syncthreads();                         // all reads of mbuf done
  {
    float bb = b2[n];
#pragma unroll
    for (int r = 0; r < 8; ++r) {
      mbuf[(mh + r) * Hn + n]      = (_Float16)gelu_exact(d0[r] + bb);
      mbuf[(16 + mh + r) * Hn + n] = (_Float16)gelu_exact(d1[r] + bb);
    }
  }
  __syncthreads();

  // ---- layer 3 : 32x128 @ 128x128, masked sum over K rows ----
  v8f e0 = vzero, e1 = vzero;
  for (int ks = 0; ks < 4; ++ks) {
    v16h bf = *(const v16h*)(W3p + ((size_t)(ks * 8 + nb) << 9) + (lane << 4));
    e0 = wmma_f16(load_a_frag(mbuf, Hn, 0, ks, lane), bf, e0);
    e1 = wmma_f16(load_a_frag(mbuf, Hn, 1, ks, lane), bf, e1);
  }
  {
    float bb   = b3[n];
    float part = 0.0f;
    const float* mrow = maskA + (size_t)bl * Kn;
#pragma unroll
    for (int r = 0; r < 8; ++r) {
      part += mrow[mh + r]      * (e0[r] + bb);
      part += mrow[16 + mh + r] * (e1[r] + bb);
    }
    atomicAdd(&colsum[n], part);           // ds_add_f32, column sum over K
  }
  __syncthreads();

  // ---- residual + LayerNorm1 ----
  float x = 0.0f;
  if (tid < Hn) {
    x = hv_row[tid] + colsum[tid] * SCALE_INV;
    red[tid] = x; red2[tid] = x * x;
  }
  __syncthreads();
  for (int s = Hn >> 1; s > 0; s >>= 1) {
    if (tid < s) { red[tid] += red[tid + s]; red2[tid] += red2[tid + s]; }
    __syncthreads();
  }
  if (tid < Hn) {
    float mean = red[0] * (1.0f / Hn);
    float var  = red2[0] * (1.0f / Hn) - mean * mean;
    hV1[(size_t)bl * Hn + tid] =
        (x - mean) * rsqrtf(var + EPSF) * ln1g[tid] + ln1b[tid];
  }
}

// ---------------------------------------------------------------------------
// Kernel 2: position-wise FFN 128 -> 512 -> 128, residual + LN2 + mask_V.
// 32 nodes per 256-thread WG.
// ---------------------------------------------------------------------------
__global__ __launch_bounds__(256) void ffn_ln2(
    const float* __restrict__ hV1, const float* __restrict__ maskV,
    const _Float16* __restrict__ Wip, const float* __restrict__ bi,
    const _Float16* __restrict__ Wop, const float* __restrict__ bo,
    const float* __restrict__ ln2g, const float* __restrict__ ln2b,
    float* __restrict__ hVout) {
  __shared__ _Float16 abuf[32 * Hn];      //  8 KB
  __shared__ _Float16 hbuf[32 * FFn];     // 32 KB
  __shared__ float    xbuf[32 * Hn];      // 16 KB
  __shared__ float    rm[32], rv[32];

  const int node0 = blockIdx.x * 32;
  const int tid = threadIdx.x, lane = tid & 31, wave = tid >> 5;
  const v8f vzero = {0, 0, 0, 0, 0, 0, 0, 0};

  for (int e = tid; e < 32 * Hn; e += 256)
    abuf[e] = (_Float16)hV1[(size_t)node0 * Hn + e];
  __syncthreads();

  const int mh = (lane >> 4) ? 8 : 0;

  // ---- hidden: 32x128 @ 128x512, GELU ----
  for (int q = 0; q < 4; ++q) {
    int nb = wave * 4 + q;
    int n  = nb * 16 + (lane & 15);
    v8f c0 = vzero, c1 = vzero;
    for (int ks = 0; ks < 4; ++ks) {
      v16h bf = *(const v16h*)(Wip + ((size_t)(ks * 32 + nb) << 9) + (lane << 4));
      c0 = wmma_f16(load_a_frag(abuf, Hn, 0, ks, lane), bf, c0);
      c1 = wmma_f16(load_a_frag(abuf, Hn, 1, ks, lane), bf, c1);
    }
    float bb = bi[n];
#pragma unroll
    for (int r = 0; r < 8; ++r) {
      hbuf[(mh + r) * FFn + n]      = (_Float16)gelu_exact(c0[r] + bb);
      hbuf[(16 + mh + r) * FFn + n] = (_Float16)gelu_exact(c1[r] + bb);
    }
  }
  __syncthreads();

  // ---- output: 32x512 @ 512x128, + bo + residual ----
  {
    int nb = wave;
    int n  = nb * 16 + (lane & 15);
    v8f d0 = vzero, d1 = vzero;
    for (int ks = 0; ks < 16; ++ks) {
      v16h bf = *(const v16h*)(Wop + ((size_t)(ks * 8 + nb) << 9) + (lane << 4));
      d0 = wmma_f16(load_a_frag(hbuf, FFn, 0, ks, lane), bf, d0);
      d1 = wmma_f16(load_a_frag(hbuf, FFn, 1, ks, lane), bf, d1);
    }
    float bb = bo[n];
#pragma unroll
    for (int r = 0; r < 8; ++r) {
      int r0 = mh + r, r1 = 16 + mh + r;
      xbuf[r0 * Hn + n] = d0[r] + bb + hV1[(size_t)(node0 + r0) * Hn + n];
      xbuf[r1 * Hn + n] = d1[r] + bb + hV1[(size_t)(node0 + r1) * Hn + n];
    }
  }
  __syncthreads();

  // ---- LN2 + mask_V ----
  if (tid < 32) {
    float s = 0.0f, s2 = 0.0f;
    for (int c = 0; c < Hn; ++c) { float v = xbuf[tid * Hn + c]; s += v; s2 += v * v; }
    float mean = s * (1.0f / Hn);
    rm[tid] = mean;
    rv[tid] = s2 * (1.0f / Hn) - mean * mean;
  }
  __syncthreads();
  for (int e = tid; e < 32 * Hn; e += 256) {
    int r = e >> 7, c = e & (Hn - 1);
    float y = (xbuf[e] - rm[r]) * rsqrtf(rv[r] + EPSF) * ln2g[c] + ln2b[c];
    hVout[(size_t)node0 * Hn + e] = y * maskV[node0 + r];
  }
}

// ---------------------------------------------------------------------------
// Kernel 3: edge-update MLP (uses updated h_V), per-edge residual + LN3.
// ---------------------------------------------------------------------------
__global__ __launch_bounds__(256) void edge_msg_ln3(
    const float* __restrict__ hVnew, const float* __restrict__ hE,
    const int* __restrict__ Eidx,
    const _Float16* __restrict__ W11p, const float* __restrict__ b11,
    const _Float16* __restrict__ W12p, const float* __restrict__ b12,
    const _Float16* __restrict__ W13p, const float* __restrict__ b13,
    const float* __restrict__ ln3g, const float* __restrict__ ln3b,
    float* __restrict__ hEout) {
  __shared__ _Float16 abuf[32 * ACOLS];   // 24 KB
  __shared__ _Float16 mbuf[32 * Hn];      //  8 KB
  __shared__ float    xbuf[32 * Hn];      // 16 KB
  __shared__ float    rm[32], rv[32];

  const int bl   = blockIdx.x;
  const int b    = bl >> 11;
  const int tid  = threadIdx.x;
  const int lane = tid & 31, wave = tid >> 5;
  const v8f vzero = {0, 0, 0, 0, 0, 0, 0, 0};

  const float* hv_row = hVnew + (size_t)bl * Hn;
  const float* he_row = hE + (size_t)bl * Kn * Hn;
  const int*   ix_row = Eidx + (size_t)bl * Kn;

  for (int e = tid; e < 32 * ACOLS; e += 256) {
    int r = e / ACOLS, c = e - r * ACOLS;
    float v;
    if (c < Hn)          v = hv_row[c];
    else if (c < 2 * Hn) v = he_row[r * Hn + (c - Hn)];
    else                 v = hVnew[((size_t)b * Ln + ix_row[r]) * Hn + (c - 2 * Hn)];
    abuf[e] = (_Float16)v;
  }
  __syncthreads();

  const int nb = wave;
  const int n  = nb * 16 + (lane & 15);
  const int mh = (lane >> 4) ? 8 : 0;

  v8f c0 = vzero, c1 = vzero;
  for (int ks = 0; ks < 12; ++ks) {
    v16h bf = *(const v16h*)(W11p + ((size_t)(ks * 8 + nb) << 9) + (lane << 4));
    c0 = wmma_f16(load_a_frag(abuf, ACOLS, 0, ks, lane), bf, c0);
    c1 = wmma_f16(load_a_frag(abuf, ACOLS, 1, ks, lane), bf, c1);
  }
  {
    float bb = b11[n];
#pragma unroll
    for (int r = 0; r < 8; ++r) {
      mbuf[(mh + r) * Hn + n]      = (_Float16)gelu_exact(c0[r] + bb);
      mbuf[(16 + mh + r) * Hn + n] = (_Float16)gelu_exact(c1[r] + bb);
    }
  }
  __syncthreads();

  v8f d0 = vzero, d1 = vzero;
  for (int ks = 0; ks < 4; ++ks) {
    v16h bf = *(const v16h*)(W12p + ((size_t)(ks * 8 + nb) << 9) + (lane << 4));
    d0 = wmma_f16(load_a_frag(mbuf, Hn, 0, ks, lane), bf, d0);
    d1 = wmma_f16(load_a_frag(mbuf, Hn, 1, ks, lane), bf, d1);
  }
  __syncthreads();
  {
    float bb = b12[n];
#pragma unroll
    for (int r = 0; r < 8; ++r) {
      mbuf[(mh + r) * Hn + n]      = (_Float16)gelu_exact(d0[r] + bb);
      mbuf[(16 + mh + r) * Hn + n] = (_Float16)gelu_exact(d1[r] + bb);
    }
  }
  __syncthreads();

  v8f e0 = vzero, e1 = vzero;
  for (int ks = 0; ks < 4; ++ks) {
    v16h bf = *(const v16h*)(W13p + ((size_t)(ks * 8 + nb) << 9) + (lane << 4));
    e0 = wmma_f16(load_a_frag(mbuf, Hn, 0, ks, lane), bf, e0);
    e1 = wmma_f16(load_a_frag(mbuf, Hn, 1, ks, lane), bf, e1);
  }
  {
    float bb = b13[n];
#pragma unroll
    for (int r = 0; r < 8; ++r) {
      int r0 = mh + r, r1 = 16 + mh + r;
      xbuf[r0 * Hn + n] = he_row[r0 * Hn + n] + e0[r] + bb;
      xbuf[r1 * Hn + n] = he_row[r1 * Hn + n] + e1[r] + bb;
    }
  }
  __syncthreads();

  if (tid < 32) {
    float s = 0.0f, s2 = 0.0f;
    for (int c = 0; c < Hn; ++c) { float v = xbuf[tid * Hn + c]; s += v; s2 += v * v; }
    float mean = s * (1.0f / Hn);
    rm[tid] = mean;
    rv[tid] = s2 * (1.0f / Hn) - mean * mean;
  }
  __syncthreads();
  for (int e = tid; e < 32 * Hn; e += 256) {
    int r = e >> 7, c = e & (Hn - 1);
    float y = (xbuf[e] - rm[r]) * rsqrtf(rv[r] + EPSF) * ln3g[c] + ln3b[c];
    hEout[(size_t)bl * Kn * Hn + e] = y;
  }
}

// ---------------------------------------------------------------------------
extern "C" void kernel_launch(void* const* d_in, const int* in_sizes, int n_in,
                              void* d_out, int out_size, void* d_ws, size_t ws_size,
                              hipStream_t stream) {
  const float* hV    = (const float*)d_in[0];
  const float* hE    = (const float*)d_in[1];
  const int*   Eidx  = (const int*)d_in[2];
  const float* maskV = (const float*)d_in[3];
  const float* maskA = (const float*)d_in[4];
  const float* W1  = (const float*)d_in[5];  const float* b1  = (const float*)d_in[6];
  const float* W2  = (const float*)d_in[7];  const float* b2  = (const float*)d_in[8];
  const float* W3  = (const float*)d_in[9];  const float* b3  = (const float*)d_in[10];
  const float* W11 = (const float*)d_in[11]; const float* b11 = (const float*)d_in[12];
  const float* W12 = (const float*)d_in[13]; const float* b12 = (const float*)d_in[14];
  const float* W13 = (const float*)d_in[15]; const float* b13 = (const float*)d_in[16];
  const float* Wi  = (const float*)d_in[17]; const float* bi  = (const float*)d_in[18];
  const float* Wo  = (const float*)d_in[19]; const float* bo  = (const float*)d_in[20];
  const float* ln1g = (const float*)d_in[21]; const float* ln1b = (const float*)d_in[22];
  const float* ln2g = (const float*)d_in[23]; const float* ln2b = (const float*)d_in[24];
  const float* ln3g = (const float*)d_in[25]; const float* ln3b = (const float*)d_in[26];

  // workspace layout (halves): packed f16 weights, then hV1 (f32)
  _Float16* wsh  = (_Float16*)d_ws;
  _Float16* W1p  = wsh + 0;        // 384x128 -> 49152
  _Float16* W2p  = wsh + 49152;    // 128x128 -> 16384
  _Float16* W3p  = wsh + 65536;    // 16384
  _Float16* W11p = wsh + 81920;    // 49152
  _Float16* W12p = wsh + 131072;   // 16384
  _Float16* W13p = wsh + 147456;   // 16384
  _Float16* Wip  = wsh + 163840;   // 128x512 -> 65536
  _Float16* Wop  = wsh + 229376;   // 512x128 -> 65536 (ends 294912)
  float* hV1 = (float*)((char*)d_ws + (size_t)294912 * sizeof(_Float16));

  float* hVout = (float*)d_out;
  float* hEout = hVout + (size_t)Bn * Ln * Hn;

  auto pack = [&](const float* s, _Float16* d, int Kd, int Nd) {
    int tot = Kd * Nd;
    pack_weight<<<(tot + 255) / 256, 256, 0, stream>>>(s, d, Kd, Nd);
  };
  pack(W1,  W1p,  ACOLS, Hn);
  pack(W2,  W2p,  Hn,    Hn);
  pack(W3,  W3p,  Hn,    Hn);
  pack(W11, W11p, ACOLS, Hn);
  pack(W12, W12p, Hn,    Hn);
  pack(W13, W13p, Hn,    Hn);
  pack(Wi,  Wip,  Hn,    FFn);
  pack(Wo,  Wop,  FFn,   Hn);

  node_msg_ln1<<<Bn * Ln, 256, 0, stream>>>(hV, hE, Eidx, maskA,
                                            W1p, b1, W2p, b2, W3p, b3,
                                            ln1g, ln1b, hV1);
  ffn_ln2<<<(Bn * Ln) / 32, 256, 0, stream>>>(hV1, maskV, Wip, bi, Wop, bo,
                                              ln2g, ln2b, hVout);
  edge_msg_ln3<<<Bn * Ln, 256, 0, stream>>>(hVout, hE, Eidx,
                                            W11p, b11, W12p, b12, W13p, b13,
                                            ln3g, ln3b, hEout);
}
